// GraphConvolution_52381421142755
// MI455X (gfx1250) — compile-verified, matrix operand order checked by reading, not run on hardware
//
#include <hip/hip_runtime.h>
#include <hip/hip_bf16.h>
#include <math.h>

#define NN 4096
#define DIN 1024
#define DOUT 512
#define LRELU 0.2f

typedef __attribute__((ext_vector_type(16))) __bf16        v16bf;
typedef __attribute__((ext_vector_type(8)))  float         v8f;
typedef __attribute__((ext_vector_type(4)))  unsigned int  u32x4;

union Frag16 { v16bf v; unsigned short u[16]; u32x4 q[2]; };

__device__ __forceinline__ unsigned short f32_to_bf16(float f) {
    unsigned int u = __float_as_uint(f);
    u += 0x7FFFu + ((u >> 16) & 1u);   // round-to-nearest-even
    return (unsigned short)(u >> 16);
}

__device__ __forceinline__ float lrelu(float e) { return e >= 0.f ? e : LRELU * e; }

// ---------------- Kernel 1: v1 = Wf @ a1, v2 = Wf @ a2 -----------------------
__global__ __launch_bounds__(256) void gat_fold_kernel(
    const float* __restrict__ Wf, const float* __restrict__ a1,
    const float* __restrict__ a2, float* __restrict__ v1, float* __restrict__ v2) {
    int k = blockIdx.x * 256 + threadIdx.x;
    if (k >= DIN) return;
    const float* wr = Wf + (size_t)k * DOUT;
    float s1 = 0.f, s2 = 0.f;
    for (int o = 0; o < DOUT; ++o) { float w = wr[o]; s1 += w * a1[o]; s2 += w * a2[o]; }
    v1[k] = s1; v2[k] = s2;
}

// ---------------- Kernel 2: f1 = x@v1 + b1, f2 = x@v2 + b2 -------------------
__global__ __launch_bounds__(256) void gat_f12_kernel(
    const float* __restrict__ x, const float* __restrict__ v1, const float* __restrict__ v2,
    const float* __restrict__ b1, const float* __restrict__ b2,
    float* __restrict__ f1, float* __restrict__ f2) {
    int row = blockIdx.x, tid = threadIdx.x;
    const float* xr = x + (size_t)row * DIN;
    float s1 = 0.f, s2 = 0.f;
    for (int k = tid; k < DIN; k += 256) { float xv = xr[k]; s1 += xv * v1[k]; s2 += xv * v2[k]; }
    __shared__ float sh1[256], sh2[256];
    sh1[tid] = s1; sh2[tid] = s2; __syncthreads();
    for (int off = 128; off > 0; off >>= 1) {
        if (tid < off) { sh1[tid] += sh1[tid + off]; sh2[tid] += sh2[tid + off]; }
        __syncthreads();
    }
    if (tid == 0) { f1[row] = sh1[0] + b1[0]; f2[row] = sh2[0] + b2[0]; }
}

// ---------------- Kernel 3: per-row online softmax stats ---------------------
__global__ __launch_bounds__(256) void gat_rowstats_kernel(
    const float* __restrict__ adj, const float* __restrict__ f1, const float* __restrict__ f2,
    float* __restrict__ mr, float* __restrict__ zr, float* __restrict__ pd) {
    int row = blockIdx.x, tid = threadIdx.x;
    float f1i = f1[row];
    const float* ar = adj + (size_t)row * NN;
    float m = -INFINITY, s = 0.f;
    for (int j = tid; j < NN; j += 256) {
        float a = ar[j];
        if (a != 0.f) {
            float e = lrelu(f1i + f2[j]);
            if (e > m) { s = s * __expf(m - e) + 1.f; m = e; }  // m=-inf => exp()=0
            else        { s += __expf(e - m); }
        }
    }
    __shared__ float sm[256], ss[256];
    sm[tid] = m; ss[tid] = s; __syncthreads();
    for (int off = 128; off > 0; off >>= 1) {
        if (tid < off) {
            float m1 = sm[tid], s1 = ss[tid], m2 = sm[tid + off], s2 = ss[tid + off];
            float mm = fmaxf(m1, m2);
            float sn = (mm == -INFINITY) ? 0.f : s1 * __expf(m1 - mm) + s2 * __expf(m2 - mm);
            sm[tid] = mm; ss[tid] = sn;
        }
        __syncthreads();
    }
    if (tid == 0) {
        float mm = sm[0], z = ss[0];                 // self-loop => mm finite, z >= 1
        float eii = lrelu(f1i + f2[row]);
        mr[row] = mm; zr[row] = z;
        pd[row] = __expf(eii - mm) / z;
    }
}

// ---------------- Kernel 4: colsum_P via symmetry; s_i factor ----------------
__global__ __launch_bounds__(256) void gat_colsum_kernel(
    const float* __restrict__ adj, const float* __restrict__ f1, const float* __restrict__ f2,
    const float* __restrict__ mr, const float* __restrict__ zr,
    const float* __restrict__ pd, float* __restrict__ sf) {
    int row = blockIdx.x, tid = threadIdx.x;     // adj symmetric: adj[i][j]==adj[j][i]
    float f2i = f2[row];
    const float* ar = adj + (size_t)row * NN;
    float acc = 0.f;
    for (int j = tid; j < NN; j += 256) {
        float a = ar[j];
        if (a != 0.f) {
            float e = lrelu(f1[j] + f2i);        // P[j,i] numerator
            acc += __expf(e - mr[j]) / zr[j];
        }
    }
    __shared__ float sh[256];
    sh[tid] = acc; __syncthreads();
    for (int off = 128; off > 0; off >>= 1) {
        if (tid < off) sh[tid] += sh[tid + off];
        __syncthreads();
    }
    if (tid == 0) {
        float colsum = sh[0];
        sf[row] = (pd[row] + 1.f) / (1.5f + 0.5f * colsum);   // rowsum = 1.5 + 0.5*colsum
    }
}

// ---------------- Pre-pass A: x (f32) -> x_bf16, packed pairs ----------------
__global__ __launch_bounds__(256) void gat_cvt_x_kernel(
    const float* __restrict__ x, unsigned int* __restrict__ xb /* NN*DIN/2 dwords */) {
    int t = blockIdx.x * 256 + threadIdx.x;      // one thread = 8 elements
    size_t base = (size_t)t * 8;
    float4 q0 = *(const float4*)(x + base);
    float4 q1 = *(const float4*)(x + base + 4);
    unsigned int* o = xb + (size_t)t * 4;
    o[0] = (unsigned int)f32_to_bf16(q0.x) | ((unsigned int)f32_to_bf16(q0.y) << 16);
    o[1] = (unsigned int)f32_to_bf16(q0.z) | ((unsigned int)f32_to_bf16(q0.w) << 16);
    o[2] = (unsigned int)f32_to_bf16(q1.x) | ((unsigned int)f32_to_bf16(q1.y) << 16);
    o[3] = (unsigned int)f32_to_bf16(q1.z) | ((unsigned int)f32_to_bf16(q1.w) << 16);
}

// ---------------- Pre-pass B: pack W into B-fragment layout ------------------
// pb[((kb*DOUT + c)*2 + h)*16 + t] = bf16(W[k, c]),
//   k = kb*32 + h*8 + (t<8 ? t : 8 + t)   (runs t=0..7 -> k0..k0+7, t=8..15 -> k0+16..k0+23)
__global__ __launch_bounds__(256) void gat_pack_w_kernel(
    const float* __restrict__ W, unsigned short* __restrict__ pb) {
    int t = blockIdx.x * 256 + threadIdx.x;      // 32768 threads: (kb, c, h)
    int h  = t & 1;
    int c  = (t >> 1) & (DOUT - 1);
    int kb = t >> 10;
    int kbase = kb * 32 + h * 8;
    unsigned short* o = pb + (size_t)t * 16;
#pragma unroll
    for (int i = 0; i < 8; ++i) o[i]     = f32_to_bf16(W[(size_t)(kbase + i) * DOUT + c]);
#pragma unroll
    for (int i = 0; i < 8; ++i) o[8 + i] = f32_to_bf16(W[(size_t)(kbase + 16 + i) * DOUT + c]);
}

// ---------------- Kernel 5 (fast): WMMA GEMM on pre-converted data -----------
__device__ __forceinline__ void store_tile(float* __restrict__ out, const float* __restrict__ sf,
                                           v8f c, int rowbase, int col) {
#pragma unroll
    for (int r = 0; r < 8; ++r) {
        float v = c[r] * sf[rowbase + r];
        out[(size_t)(rowbase + r) * DOUT + col] = v > 0.f ? v : 0.f;
    }
}

__global__ __launch_bounds__(256) void gat_gemm_fast_kernel(
    const unsigned short* __restrict__ xb,   // bf16 x, row-major [NN][DIN]
    const unsigned short* __restrict__ pb,   // packed bf16 B fragments
    const float* __restrict__ sf, float* __restrict__ out) {
    const int lane = threadIdx.x & 31;
    const int wave = threadIdx.x >> 5;
    const int tile = blockIdx.x * 8 + wave;           // 2048 tiles
    const int tile_m = tile >> 4;
    const int tile_n = tile & 15;
    const int row0 = tile_m * 32;
    const int col0 = tile_n * 32;
    const int mrow  = lane & 15;
    const int h     = lane >> 4;                      // lane half
    const int khalf = h << 3;                         // 0 or 8

    v8f c00 = {}, c01 = {}, c10 = {}, c11 = {};
    const unsigned short* xr0 = xb + (size_t)(row0 + mrow) * DIN + khalf;
    const unsigned short* xr1 = xb + (size_t)(row0 + 16 + mrow) * DIN + khalf;
    // B base for (kb=0, col, h): advance by 32*DOUT ushorts per kb
    const unsigned short* pb0 = pb + ((size_t)(col0 + (lane & 15)) * 2 + h) * 16;
    const unsigned short* pb1 = pb + ((size_t)(col0 + 16 + (lane & 15)) * 2 + h) * 16;

    for (int k0 = 0; k0 < DIN; k0 += 32) {
        Frag16 a0, a1, b0, b1;
        a0.q[0] = *(const u32x4*)(xr0 + k0);
        a0.q[1] = *(const u32x4*)(xr0 + k0 + 16);
        a1.q[0] = *(const u32x4*)(xr1 + k0);
        a1.q[1] = *(const u32x4*)(xr1 + k0 + 16);
        const size_t kboff = (size_t)(k0 >> 5) * (32 * DOUT);
        b0.q[0] = *(const u32x4*)(pb0 + kboff);
        b0.q[1] = *(const u32x4*)(pb0 + kboff + 8);
        b1.q[0] = *(const u32x4*)(pb1 + kboff);
        b1.q[1] = *(const u32x4*)(pb1 + kboff + 8);
        c00 = __builtin_amdgcn_wmma_f32_16x16x32_bf16(false, a0.v, false, b0.v, (short)0, c00, false, false);
        c01 = __builtin_amdgcn_wmma_f32_16x16x32_bf16(false, a0.v, false, b1.v, (short)0, c01, false, false);
        c10 = __builtin_amdgcn_wmma_f32_16x16x32_bf16(false, a1.v, false, b0.v, (short)0, c10, false, false);
        c11 = __builtin_amdgcn_wmma_f32_16x16x32_bf16(false, a1.v, false, b1.v, (short)0, c11, false, false);
    }

    const int hi8 = h << 3;                           // C/D layout: M = r + 8*(lane>=16)
    store_tile(out, sf, c00, row0 + hi8,      col0 + (lane & 15));
    store_tile(out, sf, c01, row0 + hi8,      col0 + 16 + (lane & 15));
    store_tile(out, sf, c10, row0 + 16 + hi8, col0 + (lane & 15));
    store_tile(out, sf, c11, row0 + 16 + hi8, col0 + 16 + (lane & 15));
}

// ---------------- Kernel 5 (fallback): convert-on-the-fly WMMA GEMM ----------
__device__ __forceinline__ v16bf load_row_frag(const float* __restrict__ p, int khalf) {
    Frag16 f;
    float4 q0 = *(const float4*)(p + khalf);
    float4 q1 = *(const float4*)(p + khalf + 4);
    float4 q2 = *(const float4*)(p + khalf + 16);
    float4 q3 = *(const float4*)(p + khalf + 20);
    f.u[0] = f32_to_bf16(q0.x); f.u[1] = f32_to_bf16(q0.y);
    f.u[2] = f32_to_bf16(q0.z); f.u[3] = f32_to_bf16(q0.w);
    f.u[4] = f32_to_bf16(q1.x); f.u[5] = f32_to_bf16(q1.y);
    f.u[6] = f32_to_bf16(q1.z); f.u[7] = f32_to_bf16(q1.w);
    f.u[8]  = f32_to_bf16(q2.x); f.u[9]  = f32_to_bf16(q2.y);
    f.u[10] = f32_to_bf16(q2.z); f.u[11] = f32_to_bf16(q2.w);
    f.u[12] = f32_to_bf16(q3.x); f.u[13] = f32_to_bf16(q3.y);
    f.u[14] = f32_to_bf16(q3.z); f.u[15] = f32_to_bf16(q3.w);
    return f.v;
}

__device__ __forceinline__ v16bf load_col_frag(const float* __restrict__ W, int kbase, int col) {
    Frag16 f;
#pragma unroll
    for (int t = 0; t < 8; ++t) f.u[t]     = f32_to_bf16(W[(size_t)(kbase + t) * DOUT + col]);
#pragma unroll
    for (int t = 0; t < 8; ++t) f.u[8 + t] = f32_to_bf16(W[(size_t)(kbase + 16 + t) * DOUT + col]);
    return f.v;
}

__global__ __launch_bounds__(256) void gat_gemm_fallback_kernel(
    const float* __restrict__ x, const float* __restrict__ W,
    const float* __restrict__ sf, float* __restrict__ out) {
    const int lane = threadIdx.x & 31;
    const int wave = threadIdx.x >> 5;
    const int tile = blockIdx.x * 8 + wave;
    const int tile_m = tile >> 4;
    const int tile_n = tile & 15;
    const int row0 = tile_m * 32;
    const int col0 = tile_n * 32;
    const int mrow  = lane & 15;
    const int khalf = (lane >> 4) << 3;

    v8f c00 = {}, c01 = {}, c10 = {}, c11 = {};
    const float* xr0 = x + (size_t)(row0 + mrow) * DIN;
    const float* xr1 = x + (size_t)(row0 + 16 + mrow) * DIN;
    const int cb0 = col0 + (lane & 15);
    const int cb1 = col0 + 16 + (lane & 15);

    for (int k0 = 0; k0 < DIN; k0 += 32) {
        v16bf a0 = load_row_frag(xr0 + k0, khalf);
        v16bf a1 = load_row_frag(xr1 + k0, khalf);
        v16bf b0 = load_col_frag(W, k0 + khalf, cb0);
        v16bf b1 = load_col_frag(W, k0 + khalf, cb1);
        c00 = __builtin_amdgcn_wmma_f32_16x16x32_bf16(false, a0, false, b0, (short)0, c00, false, false);
        c01 = __builtin_amdgcn_wmma_f32_16x16x32_bf16(false, a0, false, b1, (short)0, c01, false, false);
        c10 = __builtin_amdgcn_wmma_f32_16x16x32_bf16(false, a1, false, b0, (short)0, c10, false, false);
        c11 = __builtin_amdgcn_wmma_f32_16x16x32_bf16(false, a1, false, b1, (short)0, c11, false, false);
    }

    const int hi8 = (lane >> 4) << 3;
    store_tile(out, sf, c00, row0 + hi8,      col0 + (lane & 15));
    store_tile(out, sf, c01, row0 + hi8,      col0 + 16 + (lane & 15));
    store_tile(out, sf, c10, row0 + 16 + hi8, col0 + (lane & 15));
    store_tile(out, sf, c11, row0 + 16 + hi8, col0 + 16 + (lane & 15));
}

extern "C" void kernel_launch(void* const* d_in, const int* in_sizes, int n_in,
                              void* d_out, int out_size, void* d_ws, size_t ws_size,
                              hipStream_t stream) {
    const float* x   = (const float*)d_in[0];
    const float* adj = (const float*)d_in[1];
    const float* Wf  = (const float*)d_in[2];
    const float* a1  = (const float*)d_in[3];
    const float* b1  = (const float*)d_in[4];
    const float* a2  = (const float*)d_in[5];
    const float* b2  = (const float*)d_in[6];
    const float* W   = (const float*)d_in[7];
    float* out = (float*)d_out;

    float* ws = (float*)d_ws;
    float* v1 = ws;                    // 1024
    float* v2 = ws + 1024;             // 1024
    float* f1 = ws + 2048;             // 4096
    float* f2 = ws + 6144;             // 4096
    float* mr = ws + 10240;            // 4096
    float* zr = ws + 14336;            // 4096
    float* pd = ws + 18432;            // 4096
    float* sf = ws + 22528;            // 4096  (ends at 106496 B)

    const size_t XB_OFF = 131072;                      // 128 KB aligned
    const size_t XB_BYTES = (size_t)NN * DIN * 2;      // 8 MB
    const size_t PB_OFF = XB_OFF + XB_BYTES;
    const size_t PB_BYTES = (size_t)DIN * DOUT * 2;    // 1 MB
    const bool fast = ws_size >= PB_OFF + PB_BYTES;

    gat_fold_kernel<<<dim3(4),      dim3(256), 0, stream>>>(Wf, a1, a2, v1, v2);
    gat_f12_kernel<<<dim3(NN),      dim3(256), 0, stream>>>(x, v1, v2, b1, b2, f1, f2);
    gat_rowstats_kernel<<<dim3(NN), dim3(256), 0, stream>>>(adj, f1, f2, mr, zr, pd);
    gat_colsum_kernel<<<dim3(NN),   dim3(256), 0, stream>>>(adj, f1, f2, mr, zr, pd, sf);

    if (fast) {
        unsigned int*   xb = (unsigned int*)((char*)d_ws + XB_OFF);
        unsigned short* pbuf = (unsigned short*)((char*)d_ws + PB_OFF);
        gat_cvt_x_kernel <<<dim3(NN * DIN / 8 / 256), dim3(256), 0, stream>>>(x, xb);
        gat_pack_w_kernel<<<dim3(DIN / 32 * DOUT * 2 / 256), dim3(256), 0, stream>>>(W, pbuf);
        gat_gemm_fast_kernel<<<dim3(256), dim3(256), 0, stream>>>(
            (const unsigned short*)xb, pbuf, sf, out);
    } else {
        gat_gemm_fallback_kernel<<<dim3(256), dim3(256), 0, stream>>>(x, W, sf, out);
    }
}
